// NodeEdgeBlock_22660247454217
// MI455X (gfx1250) — matrix operand order repaired and checked by reading, not compiled, not run on hardware
//
#include <hip/hip_runtime.h>

typedef __attribute__((ext_vector_type(16))) _Float16 v16h;
typedef __attribute__((ext_vector_type(8)))  _Float16 v8h;
typedef __attribute__((ext_vector_type(8)))  float    v8f;

#define Bdim 4
#define Nn   256
#define XD   256
#define ED   128
#define INVSQ 0.17677669529663687f   // 1/sqrt(32)

// ------------------------------------------------------------------
// Kernel 1: Q,K,V projections.  Q is pre-scaled by 1/sqrt(DF).
// ------------------------------------------------------------------
__global__ void __launch_bounds__(256) neb_qkv_kernel(
    const float* __restrict__ x,
    const float* __restrict__ Wq, const float* __restrict__ bq,
    const float* __restrict__ Wk, const float* __restrict__ bk,
    const float* __restrict__ Wv, const float* __restrict__ bv,
    float* __restrict__ Qs, float* __restrict__ Kp, float* __restrict__ Vp)
{
    __shared__ float xs[XD];
    const int row = blockIdx.x;
    const int c   = threadIdx.x;
    xs[c] = x[row * XD + c];
    __syncthreads();
    float aq = bq[c], ak = bk[c], av = bv[c];
    #pragma unroll 4
    for (int k = 0; k < XD; ++k) {
        const float xv = xs[k];
        aq += xv * Wq[k * XD + c];
        ak += xv * Wk[k * XD + c];
        av += xv * Wv[k * XD + c];
    }
    Qs[row * XD + c] = aq * INVSQ;
    Kp[row * XD + c] = ak;
    Vp[row * XD + c] = av;
}

// ------------------------------------------------------------------
// Kernel 2: newX = V @ Wxo + bxo   (softmax over axis=1 sums to 1 and
// V is independent of that axis -> weighted_V == V exactly).
// ------------------------------------------------------------------
__global__ void __launch_bounds__(256) neb_newx_kernel(
    const float* __restrict__ Vp,
    const float* __restrict__ Wxo, const float* __restrict__ bxo,
    float* __restrict__ outX)
{
    __shared__ float vs[XD];
    const int row = blockIdx.x;
    const int c   = threadIdx.x;
    vs[c] = Vp[row * XD + c];
    __syncthreads();
    float a = bxo[c];
    #pragma unroll 4
    for (int k = 0; k < XD; ++k) a += vs[k] * Wxo[k * XD + c];
    outX[row * XD + c] = a;
}

// ------------------------------------------------------------------
// Kernel 3: pack f32 weight [Kdim x Ndim] into f16 WMMA B-fragment
// order: block (kc,nt), lane l, half h -> W[kc*32+(h>=8?16:0)+(l>=16?8:0)+(h&7),
// nt*16+(l&15)].  Consumer loads 32 contiguous bytes per lane.
// ------------------------------------------------------------------
__global__ void __launch_bounds__(256) neb_pack_kernel(
    const float* __restrict__ W, _Float16* __restrict__ P,
    int Kdim, int Ndim)
{
    const int idx = blockIdx.x * 256 + threadIdx.x;
    if (idx >= Kdim * Ndim) return;
    const int h    = idx & 15;
    const int lane = (idx >> 4) & 31;
    const int blk  = idx >> 9;
    const int NT   = Ndim >> 4;
    const int nt   = blk % NT;
    const int kc   = blk / NT;
    const int K = kc * 32 + ((h >> 3) << 4) + ((lane >> 4) << 3) + (h & 7);
    const int n = nt * 16 + (lane & 15);
    P[idx] = (_Float16)W[(size_t)K * Ndim + n];
}

// ------------------------------------------------------------------
// Kernel 4 (main): per 16-edge tile:
//   E1/E2 = e_tile @ Wem/Wea            (WMMA f16, f32 accum)
//   Y     = Qs[i]*K[j]*(E1+bem+1)+(E2+bea)
//   newE  = Y @ Weo + beo               (WMMA f16, f32 accum)
// One wave per tile; each wave runs TPW tiles with double-buffered
// GLOBAL_LOAD_ASYNC_TO_LDS_B128 staging of the e tile (ASYNCcnt).
// ------------------------------------------------------------------
#define NWAVES 4
#define TPW    4

__global__ void __launch_bounds__(128) neb_edge_kernel(
    const float*    __restrict__ e,
    const float*    __restrict__ Qs,
    const float*    __restrict__ Kp,
    const _Float16* __restrict__ pWem,
    const _Float16* __restrict__ pWea,
    const _Float16* __restrict__ pWeo,
    const float*    __restrict__ bem,
    const float*    __restrict__ bea,
    const float*    __restrict__ beo,
    float*          __restrict__ newE)
{
    __shared__ __align__(16) float    ldsEf[NWAVES][2][16 * ED]; // 2x8KB/wave
    __shared__ __align__(16) _Float16 ldsY [NWAVES][16 * XD];    // 8KB/wave

    const int lane = threadIdx.x & 31;
    const int wave = threadIdx.x >> 5;
    const int tile0 = (blockIdx.x * NWAVES + wave) * TPW;

    unsigned ldsBase[2];
    ldsBase[0] = (unsigned)(size_t)(void*)&ldsEf[wave][0][0];
    ldsBase[1] = (unsigned)(size_t)(void*)&ldsEf[wave][1][0];
    _Float16* Yt = ldsY[wave];

    // Async-stage one 16x128 f32 e-tile (8KB) into LDS buffer `buf`:
    // 16 issues x 32 lanes x 16B.  INST_OFFSET applies to both sides,
    // but we just advance the per-lane addresses (cheap VALU adds).
    auto issue_tile = [&](int tile, int buf) {
        unsigned            la = ldsBase[buf] + (unsigned)lane * 16u;
        unsigned long long  ga = (unsigned long long)(size_t)
                                 ((const char*)(e + (size_t)tile * 16 * ED) + lane * 16);
        #pragma unroll
        for (int t = 0; t < 16; ++t) {
            asm volatile("global_load_async_to_lds_b128 %0, %1, off"
                         :: "v"(la), "v"(ga) : "memory");
            la += 512u;
            ga += 512ull;
        }
    };

    issue_tile(tile0, 0);

    const int mrow = lane & 15;            // A-frag row
    const int kh   = (lane >> 4) << 3;     // A-frag K sub-offset (0/8)
    const int cl   = lane & 15;            // C-frag column
    const int mh   = (lane >> 4) << 3;     // C-frag row offset (0/8)

    for (int tt = 0; tt < TPW; ++tt) {
        const int tile = tile0 + tt;
        const int m0   = tile << 4;
        const int b    = m0 >> 16;         // / (N*N)
        const int rem  = m0 & 65535;
        const int i    = rem >> 8;
        const int j0   = rem & 255;

        if (tt + 1 < TPW) {
            issue_tile(tile + 1, (tt + 1) & 1);
            asm volatile("s_wait_asynccnt 16" ::: "memory"); // current buf done
        } else {
            asm volatile("s_wait_asynccnt 0" ::: "memory");
        }

        // ---- build A fragments for GEMM1 from f32 LDS tile ----
        const float* Ef = &ldsEf[wave][tt & 1][0];
        v16h ae[4];
        #pragma unroll
        for (int kc = 0; kc < 4; ++kc) {
            const float4* p = (const float4*)(Ef + mrow * ED + kc * 32 + kh);
            const float4* q = (const float4*)(Ef + mrow * ED + kc * 32 + kh + 16);
            const float4 f0 = p[0], f1 = p[1], f2 = q[0], f3 = q[1];
            ae[kc][0]  = (_Float16)f0.x; ae[kc][1]  = (_Float16)f0.y;
            ae[kc][2]  = (_Float16)f0.z; ae[kc][3]  = (_Float16)f0.w;
            ae[kc][4]  = (_Float16)f1.x; ae[kc][5]  = (_Float16)f1.y;
            ae[kc][6]  = (_Float16)f1.z; ae[kc][7]  = (_Float16)f1.w;
            ae[kc][8]  = (_Float16)f2.x; ae[kc][9]  = (_Float16)f2.y;
            ae[kc][10] = (_Float16)f2.z; ae[kc][11] = (_Float16)f2.w;
            ae[kc][12] = (_Float16)f3.x; ae[kc][13] = (_Float16)f3.y;
            ae[kc][14] = (_Float16)f3.z; ae[kc][15] = (_Float16)f3.w;
        }

        const float* qrow = Qs + (size_t)((b << 8) + i) * XD;

        // ---- GEMM1 + fused elementwise, 16 N-tiles of 16 cols ----
        for (int nt = 0; nt < 16; ++nt) {
            const int cb = nt << 4;
            v8f acc1 = {0.f,0.f,0.f,0.f,0.f,0.f,0.f,0.f};
            v8f acc2 = {0.f,0.f,0.f,0.f,0.f,0.f,0.f,0.f};
            #pragma unroll
            for (int kc = 0; kc < 4; ++kc) {
                const size_t boff = (((size_t)(kc * 16 + nt) * 32) + lane) << 4;
                const v16h b1 = *(const v16h*)(pWem + boff);
                const v16h b2 = *(const v16h*)(pWea + boff);
                acc1 = __builtin_amdgcn_wmma_f32_16x16x32_f16(
                           false, ae[kc], false, b1, (short)0, acc1, false, false);
                acc2 = __builtin_amdgcn_wmma_f32_16x16x32_f16(
                           false, ae[kc], false, b2, (short)0, acc2, false, false);
            }
            const float bm = bem[cb + cl] + 1.0f;
            const float ba = bea[cb + cl];
            const float qv = qrow[cb + cl];
            #pragma unroll
            for (int r = 0; r < 8; ++r) {
                const int m  = r + mh;
                const float kv = Kp[(size_t)((b << 8) + (j0 + m)) * XD + cb + cl];
                const float y  = (qv * kv) * (acc1[r] + bm) + (acc2[r] + ba);
                Yt[m * XD + cb + cl] = (_Float16)y;
            }
        }
        asm volatile("s_wait_dscnt 0" ::: "memory");

        // ---- A fragments for GEMM2 (Y is 16x256 f16 in LDS) ----
        v16h ay[8];
        #pragma unroll
        for (int kc = 0; kc < 8; ++kc) {
            const int base = mrow * XD + kc * 32 + kh;
            const v8h lo = *(const v8h*)(Yt + base);
            const v8h hi = *(const v8h*)(Yt + base + 16);
            #pragma unroll
            for (int t = 0; t < 8; ++t) { ay[kc][t] = lo[t]; ay[kc][8 + t] = hi[t]; }
        }

        // ---- GEMM2: newE tile = Y @ Weo + beo ----
        for (int nt = 0; nt < 8; ++nt) {
            v8f acc = {0.f,0.f,0.f,0.f,0.f,0.f,0.f,0.f};
            #pragma unroll
            for (int kc = 0; kc < 8; ++kc) {
                const size_t boff = (((size_t)(kc * 8 + nt) * 32) + lane) << 4;
                const v16h bo = *(const v16h*)(pWeo + boff);
                acc = __builtin_amdgcn_wmma_f32_16x16x32_f16(
                          false, ay[kc], false, bo, (short)0, acc, false, false);
            }
            const int cc = (nt << 4) + cl;
            const float bo = beo[cc];
            #pragma unroll
            for (int r = 0; r < 8; ++r) {
                const int m = r + mh;
                newE[(size_t)(m0 + m) * ED + cc] = acc[r] + bo;
            }
        }
    }
}

// ------------------------------------------------------------------
extern "C" void kernel_launch(void* const* d_in, const int* in_sizes, int n_in,
                              void* d_out, int out_size, void* d_ws, size_t ws_size,
                              hipStream_t stream)
{
    const float* x   = (const float*)d_in[0];
    const float* e   = (const float*)d_in[1];
    const float* Wq  = (const float*)d_in[2];
    const float* bq  = (const float*)d_in[3];
    const float* Wk  = (const float*)d_in[4];
    const float* bk  = (const float*)d_in[5];
    const float* Wv  = (const float*)d_in[6];
    const float* bv  = (const float*)d_in[7];
    const float* Wem = (const float*)d_in[8];
    const float* bem = (const float*)d_in[9];
    const float* Wea = (const float*)d_in[10];
    const float* bea = (const float*)d_in[11];
    const float* Wxo = (const float*)d_in[12];
    const float* bxo = (const float*)d_in[13];
    const float* Weo = (const float*)d_in[14];
    const float* beo = (const float*)d_in[15];

    float* outX = (float*)d_out;                           // [B,N,XD]
    float* outE = (float*)d_out + (size_t)Bdim * Nn * XD;  // [B,N,N,ED]

    char* ws = (char*)d_ws;
    float*    Qs   = (float*)(ws + 0);                  // 1 MB
    float*    Kp   = (float*)(ws + (1u << 20));         // 1 MB
    float*    Vp   = (float*)(ws + (2u << 20));         // 1 MB
    _Float16* pWem = (_Float16*)(ws + (3u << 20));                 // 64 KB
    _Float16* pWea = (_Float16*)(ws + (3u << 20) + (64u << 10));   // 64 KB
    _Float16* pWeo = (_Float16*)(ws + (3u << 20) + (128u << 10));  // 64 KB

    const int rows = Bdim * Nn;                         // 1024 node rows

    neb_qkv_kernel<<<rows, 256, 0, stream>>>(x, Wq, bq, Wk, bk, Wv, bv, Qs, Kp, Vp);
    neb_newx_kernel<<<rows, 256, 0, stream>>>(Vp, Wxo, bxo, outX);

    neb_pack_kernel<<<(ED * XD + 255) / 256, 256, 0, stream>>>(Wem, pWem, ED, XD);
    neb_pack_kernel<<<(ED * XD + 255) / 256, 256, 0, stream>>>(Wea, pWea, ED, XD);
    neb_pack_kernel<<<(XD * ED + 255) / 256, 256, 0, stream>>>(Weo, pWeo, XD, ED);

    const int tiles  = (Bdim * Nn * Nn) / 16;           // 16384
    const int blocks = tiles / (NWAVES * TPW);          // 1024
    neb_edge_kernel<<<blocks, 32 * NWAVES, 0, stream>>>(
        e, Qs, Kp, pWem, pWea, pWeo, bem, bea, beo, outE);
}
